// GPT2Attention_15882789060898
// MI455X (gfx1250) — compile-verified, matrix operand order checked by reading, not run on hardware
//
#include <hip/hip_runtime.h>
#include <math.h>

typedef __attribute__((ext_vector_type(16))) _Float16 v16h;
typedef __attribute__((ext_vector_type(8)))  _Float16 v8h;
typedef __attribute__((ext_vector_type(8)))  float    v8f;

#define S_LEN 2048
#define DMODEL 1024
#define NHEAD 16
#define HDIM 64
#define BH 32                                   // B * NHEAD
#define TOK 4096                                // B * S
#define QKV_MAT (BH * S_LEN * HDIM)             // elements per one of q/k/v

__device__ __forceinline__ v8f wmma16(v16h a, v16h b, v8f c) {
  return __builtin_amdgcn_wmma_f32_16x16x32_f16(
      /*neg_a=*/false, a, /*neg_b=*/false, b,
      /*c_mod=*/(short)0, c, /*reuse_a=*/false, /*reuse_b=*/false);
}

// Per-lane 16-half fragment slice from f16 row-major memory: two contiguous
// 16B chunks at p[0..7] and p[16..23] (ISA 16-bit A/B fragment layout).
__device__ __forceinline__ v16h load_frag_f16(const _Float16* __restrict__ p) {
  v8h lo = *(const v8h*)p;
  v8h hi = *(const v8h*)(p + 16);
  v16h r;
#pragma unroll
  for (int i = 0; i < 8; ++i) { r[i] = lo[i]; r[i + 8] = hi[i]; }
  return r;
}

// ---------------------------------------------------------------------------
// Kernel 0: one-shot f32 -> f16 conversion (X and the four weight matrices).
// ---------------------------------------------------------------------------
__global__ __launch_bounds__(256)
void cvt_f32_f16(const float* __restrict__ s, _Float16* __restrict__ d, int n) {
  const int i = (blockIdx.x * 256 + threadIdx.x) * 8;
  if (i >= n) return;
  v8h h;
#pragma unroll
  for (int j = 0; j < 8; ++j) h[j] = (_Float16)s[i + j];
  *(v8h*)(d + i) = h;
}

// ---------------------------------------------------------------------------
// Kernel 1: fused QKV projection + bias + RoPE (+ q pre-scale 1/sqrt(HD)).
// Wave tile 32M x 64N: 8 WMMAs per k-step against 12 b128 loads (B fragments
// amortized over two M tiles). N strip = one full head, so the rotate-half
// partner (hd, hd+32) sits in C-tiles (t, t+2) of the same lane.
// Outputs f16: q,k as [bh][s][hd], V TRANSPOSED as [bh][hd][s].
// ---------------------------------------------------------------------------
__global__ __launch_bounds__(256)
void qkv_rope_gemm(const _Float16* __restrict__ X,
                   const _Float16* __restrict__ Wq, const _Float16* __restrict__ Wk,
                   const _Float16* __restrict__ Wv,
                   const float* __restrict__ bq, const float* __restrict__ bk,
                   const float* __restrict__ bv,
                   _Float16* __restrict__ qh, _Float16* __restrict__ kh,
                   _Float16* __restrict__ vh) {
  const int lane = threadIdx.x & 31;
  const int wave = threadIdx.x >> 5;
  const int wm = wave & 1, wn = wave >> 1;
  const int m0 = blockIdx.x * 64 + wm * 32;
  const int n0 = blockIdx.y * 256 + wn * 64;     // multiple of 64 -> one head
  const int ml  = lane & 15;
  const int khi = (lane >> 4) << 3;

  const int which = n0 >> 10;                    // 0=q, 1=k, 2=v
  const _Float16* W  = (which == 0) ? Wq : (which == 1) ? Wk : Wv;
  const float* bias  = (which == 0) ? bq : (which == 1) ? bk : bv;
  const int nW = n0 & (DMODEL - 1);

  v8f acc[2][4] = {};
  const _Float16* arow0 = X + (size_t)(m0 + ml) * DMODEL + khi;
  const _Float16* arow1 = arow0 + (size_t)16 * DMODEL;

  for (int k0 = 0; k0 < DMODEL; k0 += 32) {
    v16h a0 = load_frag_f16(arow0 + k0);
    v16h a1 = load_frag_f16(arow1 + k0);
    __builtin_prefetch(arow0 + k0 + 256, 0, 1);
    __builtin_prefetch(W + (size_t)(nW + ml) * DMODEL + k0 + 256, 0, 1);
#pragma unroll
    for (int t = 0; t < 4; ++t) {
      v16h bf = load_frag_f16(W + (size_t)(nW + t * 16 + ml) * DMODEL + k0 + khi);
      acc[0][t] = wmma16(a0, bf, acc[0][t]);
      acc[1][t] = wmma16(a1, bf, acc[1][t]);
    }
  }

  const int h = nW >> 6;
#pragma unroll
  for (int mt = 0; mt < 2; ++mt) {
    // bias
#pragma unroll
    for (int t = 0; t < 4; ++t) {
      const float bval = bias[nW + t * 16 + ml];
#pragma unroll
      for (int r = 0; r < 8; ++r) acc[mt][t][r] += bval;
    }

    const int mrowbase = m0 + mt * 16 + ((lane >> 4) << 3);

    // RoPE on q/k in registers; scale 1/sqrt(64) folded into q.
    if (which < 2) {
      const float qscale = (which == 0) ? 0.125f : 1.0f;
#pragma unroll
      for (int t = 0; t < 2; ++t) {
        const int j = t * 16 + ml;                          // rotary index
        const float fr = exp2f((float)j * -0.41524101f);    // 10000^(-j/32)
#pragma unroll
        for (int r = 0; r < 8; ++r) {
          const int srow = (mrowbase + r) & (S_LEN - 1);
          float sn, cs;
          __sincosf((float)srow * fr, &sn, &cs);
          const float x1 = acc[mt][t][r], x2 = acc[mt][t + 2][r];
          acc[mt][t][r]     = (x1 * cs - x2 * sn) * qscale;
          acc[mt][t + 2][r] = (x2 * cs + x1 * sn) * qscale;
        }
      }
    }

    if (which == 2) {
      // V stored transposed: vh[bh][hd][s] -> PV B-fragments contiguous
#pragma unroll
      for (int t = 0; t < 4; ++t) {
        const int hd = t * 16 + ml;
#pragma unroll
        for (int r = 0; r < 8; ++r) {
          const int mrow = mrowbase + r;
          const int b = mrow >> 11, s = mrow & (S_LEN - 1);
          vh[((size_t)(b * NHEAD + h) * HDIM + hd) * S_LEN + s] =
              (_Float16)acc[mt][t][r];
        }
      }
    } else {
      _Float16* dst = (which == 0) ? qh : kh;
#pragma unroll
      for (int t = 0; t < 4; ++t) {
        const int hd = t * 16 + ml;
#pragma unroll
        for (int r = 0; r < 8; ++r) {
          const int mrow = mrowbase + r;
          const int b = mrow >> 11, s = mrow & (S_LEN - 1);
          dst[((size_t)(b * NHEAD + h) * S_LEN + s) * HDIM + hd] =
              (_Float16)acc[mt][t][r];
        }
      }
    }
  }
}

// ---------------------------------------------------------------------------
// Kernel 2: causal flash attention. One wave per (head-batch, 16-query block).
// All operands f16; P relaid C-layout -> A-layout via per-wave LDS staging.
// Output f16 in [B, S, D] layout for the final projection.
// ---------------------------------------------------------------------------
__global__ __launch_bounds__(128)
void flash_attn(const _Float16* __restrict__ qh, const _Float16* __restrict__ kh,
                const _Float16* __restrict__ vh, _Float16* __restrict__ aout) {
  __shared__ __align__(16) _Float16 pshm[4][16 * 32];
  const int lane = threadIdx.x & 31;
  const int wave = threadIdx.x >> 5;
  const int gw = blockIdx.x * 4 + wave;
  const int qb = gw & 127;                       // S/16 query blocks
  const int bh = gw >> 7;                        // 0..31
  const int m0 = qb * 16;
  const int ml  = lane & 15;
  const int khi = (lane >> 4) << 3;

  const _Float16* qbuf = qh + (size_t)bh * S_LEN * HDIM;
  const _Float16* kbuf = kh + (size_t)bh * S_LEN * HDIM;
  const _Float16* vbuf = vh + (size_t)bh * HDIM * S_LEN;   // [hd][s]
  _Float16* pl = pshm[wave];

  // resident Q fragments (HD=64 -> two K=32 slices); q pre-scaled by 1/8
  v16h qf[2];
#pragma unroll
  for (int j = 0; j < 2; ++j)
    qf[j] = load_frag_f16(qbuf + (size_t)(m0 + ml) * HDIM + j * 32 + khi);

  v8f o[4] = {};
  float mrow[8], lrow[8];
#pragma unroll
  for (int r = 0; r < 8; ++r) { mrow[r] = -3.0e38f; lrow[r] = 0.0f; }

  const int mbase = m0 + ((lane >> 4) << 3);
  const int kend = m0 + 16;                      // exclusive causal key bound

  for (int j0 = 0; j0 < kend; j0 += 32) {
    // ---- scores: two 16x16 tiles over 32 keys ----
    v8f sc[2];
#pragma unroll
    for (int t = 0; t < 2; ++t) {
      v8f z = {};
      v16h kf0 = load_frag_f16(kbuf + (size_t)(j0 + t * 16 + ml) * HDIM + khi);
      z = wmma16(qf[0], kf0, z);
      v16h kf1 = load_frag_f16(kbuf + (size_t)(j0 + t * 16 + ml) * HDIM + 32 + khi);
      sc[t] = wmma16(qf[1], kf1, z);
    }

    // ---- causal mask (only the <=2 diagonal blocks actually cut) ----
    if (j0 + 31 >= m0) {
#pragma unroll
      for (int t = 0; t < 2; ++t) {
        const int kcol = j0 + t * 16 + ml;
#pragma unroll
        for (int r = 0; r < 8; ++r)
          if (kcol > mbase + r) sc[t][r] = -3.0e38f;
      }
    }

    // ---- online softmax: row reductions across 16-lane halves ----
    float alpha[8];
#pragma unroll
    for (int r = 0; r < 8; ++r) {
      float mx = fmaxf(sc[0][r], sc[1][r]);
#pragma unroll
      for (int off = 8; off; off >>= 1)
        mx = fmaxf(mx, __shfl_xor(mx, off, 16));
      const float mnew = fmaxf(mrow[r], mx);
      alpha[r] = __expf(mrow[r] - mnew);
      mrow[r] = mnew;
      const float p0 = __expf(sc[0][r] - mnew);
      const float p1 = __expf(sc[1][r] - mnew);
      sc[0][r] = p0; sc[1][r] = p1;
      float ls = p0 + p1;
#pragma unroll
      for (int off = 8; off; off >>= 1)
        ls += __shfl_xor(ls, off, 16);
      lrow[r] = lrow[r] * alpha[r] + ls;
    }

    // rescale running output
#pragma unroll
    for (int t = 0; t < 4; ++t)
#pragma unroll
      for (int r = 0; r < 8; ++r) o[t][r] *= alpha[r];

    // ---- P: C-layout -> LDS [16][32] f16 -> A-fragment ----
#pragma unroll
    for (int t = 0; t < 2; ++t)
#pragma unroll
      for (int r = 0; r < 8; ++r)
        pl[(r + ((lane >> 4) << 3)) * 32 + t * 16 + ml] = (_Float16)sc[t][r];
    // DS ops of one wave complete in order; compiler inserts s_wait_dscnt
    v16h pf = load_frag_f16(pl + ml * 32 + khi);

    // ---- accumulate O += P @ V, V^T fragments contiguous along s ----
#pragma unroll
    for (int t = 0; t < 4; ++t) {
      v16h vf = load_frag_f16(vbuf + (size_t)(t * 16 + ml) * S_LEN + j0 + khi);
      o[t] = wmma16(pf, vf, o[t]);
    }
  }

  // ---- finalize and store f16 to [B, S, D] ----
  const int b = bh >> 4, h = bh & 15;
#pragma unroll
  for (int r = 0; r < 8; ++r) {
    const float inv = 1.0f / lrow[r];
    const int srow = m0 + ((lane >> 4) << 3) + r;
    _Float16* op = aout + ((size_t)(b * S_LEN + srow)) * DMODEL + h * HDIM;
#pragma unroll
    for (int t = 0; t < 4; ++t) op[t * 16 + ml] = (_Float16)(o[t][r] * inv);
  }
}

// ---------------------------------------------------------------------------
// Kernel 3: output projection  out = attn @ Wo^T + bo  (f16 in, f32 out)
// Wave tile 32M x 64N, same operand-reuse scheme as kernel 1.
// ---------------------------------------------------------------------------
__global__ __launch_bounds__(256)
void out_proj_gemm(const _Float16* __restrict__ A, const _Float16* __restrict__ Wo,
                   const float* __restrict__ bo, float* __restrict__ out) {
  const int lane = threadIdx.x & 31;
  const int wave = threadIdx.x >> 5;
  const int wm = wave & 1, wn = wave >> 1;
  const int m0 = blockIdx.x * 64 + wm * 32;
  const int n0 = blockIdx.y * 256 + wn * 64;
  const int ml  = lane & 15;
  const int khi = (lane >> 4) << 3;

  v8f acc[2][4] = {};
  const _Float16* arow0 = A + (size_t)(m0 + ml) * DMODEL + khi;
  const _Float16* arow1 = arow0 + (size_t)16 * DMODEL;

  for (int k0 = 0; k0 < DMODEL; k0 += 32) {
    v16h a0 = load_frag_f16(arow0 + k0);
    v16h a1 = load_frag_f16(arow1 + k0);
    __builtin_prefetch(arow0 + k0 + 256, 0, 1);
    __builtin_prefetch(Wo + (size_t)(n0 + ml) * DMODEL + k0 + 256, 0, 1);
#pragma unroll
    for (int t = 0; t < 4; ++t) {
      v16h bf = load_frag_f16(Wo + (size_t)(n0 + t * 16 + ml) * DMODEL + k0 + khi);
      acc[0][t] = wmma16(a0, bf, acc[0][t]);
      acc[1][t] = wmma16(a1, bf, acc[1][t]);
    }
  }

#pragma unroll
  for (int mt = 0; mt < 2; ++mt) {
    const int mrowbase = m0 + mt * 16 + ((lane >> 4) << 3);
#pragma unroll
    for (int t = 0; t < 4; ++t) {
      const int n = n0 + t * 16 + ml;
      const float bval = bo[n];
#pragma unroll
      for (int r = 0; r < 8; ++r)
        out[(size_t)(mrowbase + r) * DMODEL + n] = acc[mt][t][r] + bval;
    }
  }
}

extern "C" void kernel_launch(void* const* d_in, const int* in_sizes, int n_in,
                              void* d_out, int out_size, void* d_ws, size_t ws_size,
                              hipStream_t stream) {
  const float* X  = (const float*)d_in[0];
  const float* Wq = (const float*)d_in[1];
  const float* Wk = (const float*)d_in[2];
  const float* Wv = (const float*)d_in[3];
  const float* Wo = (const float*)d_in[4];
  const float* bq = (const float*)d_in[5];
  const float* bk = (const float*)d_in[6];
  const float* bv = (const float*)d_in[7];
  const float* bo = (const float*)d_in[8];

  // f16 workspace layout
  _Float16* Xh  = (_Float16*)d_ws;                         // 4194304
  _Float16* Wqh = Xh  + (size_t)TOK * DMODEL;              // 1048576 each
  _Float16* Wkh = Wqh + (size_t)DMODEL * DMODEL;
  _Float16* Wvh = Wkh + (size_t)DMODEL * DMODEL;
  _Float16* Woh = Wvh + (size_t)DMODEL * DMODEL;
  _Float16* qhb = Woh + (size_t)DMODEL * DMODEL;           // 4194304 each
  _Float16* khb = qhb + (size_t)QKV_MAT;
  _Float16* vhb = khb + (size_t)QKV_MAT;
  _Float16* ath = vhb + (size_t)QKV_MAT;                   // 4194304

  // Kernel 0: hoisted conversions (one pass each)
  cvt_f32_f16<<<dim3(TOK * DMODEL / 2048), 256, 0, stream>>>(X,  Xh,  TOK * DMODEL);
  cvt_f32_f16<<<dim3(DMODEL * DMODEL / 2048), 256, 0, stream>>>(Wq, Wqh, DMODEL * DMODEL);
  cvt_f32_f16<<<dim3(DMODEL * DMODEL / 2048), 256, 0, stream>>>(Wk, Wkh, DMODEL * DMODEL);
  cvt_f32_f16<<<dim3(DMODEL * DMODEL / 2048), 256, 0, stream>>>(Wv, Wvh, DMODEL * DMODEL);
  cvt_f32_f16<<<dim3(DMODEL * DMODEL / 2048), 256, 0, stream>>>(Wo, Woh, DMODEL * DMODEL);

  // Kernel 1: M=4096 N=3072 -> grid (4096/64, 3072/256)
  qkv_rope_gemm<<<dim3(64, 12), 256, 0, stream>>>(Xh, Wqh, Wkh, Wvh, bq, bk, bv,
                                                  qhb, khb, vhb);
  // Kernel 2: 32 head-batches * 128 query blocks = 4096 waves, 4 waves/block
  flash_attn<<<dim3(1024), 128, 0, stream>>>(qhb, khb, vhb, ath);
  // Kernel 3: M=4096 N=1024 -> grid (4096/64, 1024/256)
  out_proj_gemm<<<dim3(64, 4), 256, 0, stream>>>(ath, Woh, bo, (float*)d_out);
}
